// TrainableClusteringLoss_48610439856188
// MI455X (gfx1250) — compile-verified, matrix operand order checked by reading, not run on hardware
//
#include <hip/hip_runtime.h>
#include <hip/hip_bf16.h>

typedef __attribute__((ext_vector_type(16))) _Float16 v16h;
typedef __attribute__((ext_vector_type(8)))  _Float16 v8h;
typedef __attribute__((ext_vector_type(8)))  float    v8f;

#define N_POINTS   262144
#define N_CLUSTERS 1024
#define EMBED_DIM  512
#define N_STRIPS   (N_POINTS / 16)       // 16384 row strips, one per wave
#define WAVES_PER_BLOCK 8
#define N_CHUNKS   (EMBED_DIM / 32)      // 16 K-chunks of 32
#define N_CTILES   (N_CLUSTERS / 16)     // 64 column tiles of 16 clusters

// LDS staging: half-tile phases (8 chunks), 80B-strided records (64B data+16B pad).
// Stride 80B = 20 dwords: lane L starts at dword (20*L) mod 64 -> 16 bank
// residues, L and L+16 pair up = the 2-cycle minimum for wave32 ds_load_b128.
#define PHASE_CHUNKS 8
#define REC_STRIDE  80
#define RECS_PER_PHASE 256               // 8 chunks * 32 lanes
#define BUF_BYTES   (RECS_PER_PHASE * REC_STRIDE)   // 20480
#define PHASE_SRC_BYTES (RECS_PER_PHASE * 32)       // 8192 per hi / lo stream

// Workspace layout (bytes)
#define WS_BHI   0u                       // 1 MB: centroids hi-fp16, WMMA-B swizzled
#define WS_BLO   (1u << 20)               // 1 MB: centroids lo-fp16, WMMA-B swizzled
#define WS_CSQ   (2u << 20)               // 4 KB: ||c_k||^2
#define WS_PART  ((2u << 20) + 4096u)     // 64 KB: per-strip loss partials

// ---------------------------------------------------------------------------
// Prep 1: split centroids fp32 -> fp16 hi/lo and swizzle into the per-lane
// WMMA B-operand layout.  Record (t, c, L): cluster n = t*16 + (L&15),
// K offsets = c*32 + (L>>4)*8 + {0..7} and +16..23.
// ---------------------------------------------------------------------------
__global__ void prep_b_kernel(const float* __restrict__ cent,
                              _Float16* __restrict__ bhi,
                              _Float16* __restrict__ blo) {
    int rec = blockIdx.x * blockDim.x + threadIdx.x;        // 0 .. 32767
    int L = rec & 31;
    int c = (rec >> 5) & 15;
    int t = rec >> 9;
    int n  = t * 16 + (L & 15);
    int kb = c * 32 + (L >> 4) * 8;
    const float* src = cent + (size_t)n * EMBED_DIM + kb;
    _Float16* dh = bhi + (size_t)rec * 16;
    _Float16* dl = blo + (size_t)rec * 16;
#pragma unroll
    for (int j = 0; j < 8; ++j) {
        float f0 = src[j];
        float f1 = src[j + 16];
        _Float16 h0 = (_Float16)f0;
        _Float16 h1 = (_Float16)f1;
        dh[j]     = h0;  dl[j]     = (_Float16)(f0 - (float)h0);
        dh[j + 8] = h1;  dl[j + 8] = (_Float16)(f1 - (float)h1);
    }
}

// ---------------------------------------------------------------------------
// Prep 2: ||c_k||^2 (fp32, exact).
// ---------------------------------------------------------------------------
__global__ void prep_csq_kernel(const float* __restrict__ cent,
                                float* __restrict__ csq) {
    int n = blockIdx.x * blockDim.x + threadIdx.x;          // 0 .. 1023
    const float* row = cent + (size_t)n * EMBED_DIM;
    float s = 0.f;
    for (int j = 0; j < EMBED_DIM; ++j) s = fmaf(row[j], row[j], s);
    csq[n] = s;
}

// ---------------------------------------------------------------------------
// Main: one wave per 16-row strip; 8 waves/block share B tiles through LDS.
// Double-buffered staging (global_load -> regs -> ds_store), one barrier per
// phase; 24 WMMAs/phase in 3 independent accumulator chains.  The phase
// parity p is a compile-time constant (unrolled) so every ah[c]/al[c] index
// is constant -> A stays in (extended) VGPRs, no scratch demotion.
// ---------------------------------------------------------------------------
__global__ __launch_bounds__(256)
void assign_kernel(const float*    __restrict__ emb,
                   const _Float16* __restrict__ bhi,
                   const _Float16* __restrict__ blo,
                   const float*    __restrict__ csq,
                   float*          __restrict__ out,       // [0]=loss, [1+n]=idx
                   float*          __restrict__ partial) { // per-strip loss sums
    extern __shared__ char ldsbuf[];                       // 2 * BUF_BYTES

    const int tid    = threadIdx.x;
    const int wave   = tid >> 5;
    const int lane   = tid & 31;
    const int laneLo = lane & 15;
    const int laneHi = lane >> 4;
    const int strip  = blockIdx.x * WAVES_PER_BLOCK + wave;
    const int row0   = strip * 16;

    // ---- Load + split this strip's A tile (16 rows x 512) once -----------
    // Lane layout (ISA 16-bit A operand): lanes 0-15 row=laneLo K{0-7,16-23},
    // lanes 16-31 row=laneLo K{8-15,24-31}; v16h elem j<8 -> seg0, j>=8 -> seg1.
    v16h ah[N_CHUNKS], al[N_CHUNKS];
    float esq = 0.f;                                        // partial ||e_row||^2
    const float* arow = emb + (size_t)(row0 + laneLo) * EMBED_DIM + laneHi * 8;
#pragma unroll
    for (int c = 0; c < N_CHUNKS; ++c) {
        const float* p = arow + c * 32;
        float f[16];
        *(float4*)&f[0]  = *(const float4*)(p);
        *(float4*)&f[4]  = *(const float4*)(p + 4);
        *(float4*)&f[8]  = *(const float4*)(p + 16);
        *(float4*)&f[12] = *(const float4*)(p + 20);
        v16h h, l;
#pragma unroll
        for (int j = 0; j < 16; ++j) {
            float fj = f[j];
            esq = fmaf(fj, fj, esq);
            _Float16 hh = (_Float16)fj;
            h[j] = hh;
            l[j] = (_Float16)(fj - (float)hh);
        }
        ah[c] = h;
        al[c] = l;
    }

    // ---- B staging pipeline: thread tid owns record tid of each phase ----
    const char* ghi = (const char*)bhi + (size_t)tid * 32;
    const char* glo = (const char*)blo + (size_t)tid * 32;
    char* myslot = ldsbuf + (size_t)tid * REC_STRIDE;

    // Stage-in registers for the upcoming phase (16 VGPRs).
    float4 sh0 = *(const float4*)(ghi);
    float4 sh1 = *(const float4*)(ghi + 16);
    float4 sl0 = *(const float4*)(glo);
    float4 sl1 = *(const float4*)(glo + 16);

    // ---- Running argmin over all 1024 clusters ---------------------------
    float minv[8];
    int   mini[8];
#pragma unroll
    for (int r = 0; r < 8; ++r) { minv[r] = 3.4028235e38f; mini[r] = 0; }

    v8f acc0 = {}, acc1 = {}, acc2 = {};

    for (int t = 0; t < N_CTILES; ++t) {
#pragma unroll
        for (int p = 0; p < 2; ++p) {               // p is a COMPILE-TIME constant
            // Commit this phase's records to buf[p].  (Safe: all reads of
            // this buffer happened two phases ago, before the last barrier.)
            {
                char* slot = myslot + (size_t)p * BUF_BYTES;
                *(float4*)(slot)      = sh0;
                *(float4*)(slot + 16) = sh1;
                *(float4*)(slot + 32) = sl0;
                *(float4*)(slot + 48) = sl1;
            }

            // Issue global loads for the next phase; latency hidden under
            // this phase's 24 WMMAs.
            if (t < N_CTILES - 1 || p == 0) {
                size_t soff = (size_t)(2 * t + p + 1) * PHASE_SRC_BYTES;
                sh0 = *(const float4*)(ghi + soff);
                sh1 = *(const float4*)(ghi + soff + 16);
                sl0 = *(const float4*)(glo + soff);
                sl1 = *(const float4*)(glo + soff + 16);
            }

            __syncthreads();    // this phase's stores visible to all waves

            // Compute 8 chunks from buf[p]; records at (cl*32+lane)*80.
            const char* buf = ldsbuf + (size_t)p * BUF_BYTES
                                     + (size_t)lane * REC_STRIDE;
#pragma unroll
            for (int cl = 0; cl < PHASE_CHUNKS; ++cl) {
                const int c = p * PHASE_CHUNKS + cl;        // constant index
                const char* rp = buf + cl * (32 * REC_STRIDE);
                v8h h0 = *(const v8h*)(rp);
                v8h h1 = *(const v8h*)(rp + 16);
                v8h l0 = *(const v8h*)(rp + 32);
                v8h l1 = *(const v8h*)(rp + 48);
                v16h bh = __builtin_shufflevector(h0, h1, 0,1,2,3,4,5,6,7,
                                                          8,9,10,11,12,13,14,15);
                v16h bl = __builtin_shufflevector(l0, l1, 0,1,2,3,4,5,6,7,
                                                          8,9,10,11,12,13,14,15);
                // cross ~= eh*ch + eh*cl + el*ch : three independent chains
                acc0 = __builtin_amdgcn_wmma_f32_16x16x32_f16(
                           false, ah[c], false, bh, (short)0, acc0, false, false);
                acc1 = __builtin_amdgcn_wmma_f32_16x16x32_f16(
                           false, ah[c], false, bl, (short)0, acc1, false, false);
                acc2 = __builtin_amdgcn_wmma_f32_16x16x32_f16(
                           false, al[c], false, bh, (short)0, acc2, false, false);
            }
        }

        // Full tile accumulated: score = c_sq - 2*cross, running argmin.
        // C layout: VGPR r, lane -> row M = r + 8*laneHi, col N = laneLo.
        {
            int   col = t * 16 + laneLo;
            float cs  = csq[col];
#pragma unroll
            for (int r = 0; r < 8; ++r) {
                float cross = acc0[r] + acc1[r] + acc2[r];
                float sc = fmaf(-2.f, cross, cs);
                if (sc < minv[r]) { minv[r] = sc; mini[r] = col; } // first idx wins
            }
            acc0 = v8f{}; acc1 = v8f{}; acc2 = v8f{};
        }
    }

    // ---- Cross-lane argmin within each 16-lane half (rows r / r+8) -------
#pragma unroll
    for (int r = 0; r < 8; ++r) {
#pragma unroll
        for (int m = 1; m <= 8; m <<= 1) {
            float ov = __shfl_xor(minv[r], m, 32);
            int   oi = __shfl_xor(mini[r], m, 32);
            if (ov < minv[r] || (ov == minv[r] && oi < mini[r])) {
                minv[r] = ov; mini[r] = oi;
            }
        }
    }

    // ---- ||e_row||^2 : combine lane L and L+16 halves --------------------
    esq += __shfl_xor(esq, 16, 32);                 // lane -> e_sq[row0+laneLo]
    float esqAlt = __shfl_xor(esq, 8, 32);          // half-1 writers need row+8
    float myesq  = laneHi ? esqAlt : esq;

    // ---- Emit assignments + per-strip loss partial -----------------------
    float contrib = 0.f;
#pragma unroll
    for (int r = 0; r < 8; ++r) {
        if (laneLo == r) {                          // lanes 0-7 and 16-23 write
            int row = row0 + r + 8 * laneHi;
            out[1 + row] = (float)mini[r];
            contrib = myesq + minv[r];              // d2_min = ||e||^2 + min score
        }
    }
    contrib += __shfl_xor(contrib, 1, 32);
    contrib += __shfl_xor(contrib, 2, 32);
    contrib += __shfl_xor(contrib, 4, 32);
    contrib += __shfl_xor(contrib, 8, 32);
    contrib += __shfl_xor(contrib, 16, 32);
    if (lane == 0) partial[strip] = contrib;
}

// ---------------------------------------------------------------------------
// Deterministic final reduction: 16384 partials -> loss scalar.
// ---------------------------------------------------------------------------
__global__ void finalize_kernel(const float* __restrict__ partial,
                                float* __restrict__ out) {
    __shared__ float sm[256];
    float s = 0.f;
    for (int i = threadIdx.x; i < N_STRIPS; i += 256) s += partial[i];
    sm[threadIdx.x] = s;
    __syncthreads();
    for (int k = 128; k > 0; k >>= 1) {
        if ((int)threadIdx.x < k) sm[threadIdx.x] += sm[threadIdx.x + k];
        __syncthreads();
    }
    if (threadIdx.x == 0)
        out[0] = sm[0] / (float)((size_t)N_POINTS * EMBED_DIM);
}

// ---------------------------------------------------------------------------
extern "C" void kernel_launch(void* const* d_in, const int* in_sizes, int n_in,
                              void* d_out, int out_size, void* d_ws, size_t ws_size,
                              hipStream_t stream) {
    const float* emb  = (const float*)d_in[0];   // [262144, 512] fp32
    const float* cent = (const float*)d_in[1];   // [1024, 512]   fp32
    float* out = (float*)d_out;                  // [1 + 262144]
    char*  ws  = (char*)d_ws;

    _Float16* bhi  = (_Float16*)(ws + WS_BHI);
    _Float16* blo  = (_Float16*)(ws + WS_BLO);
    float*    csq  = (float*)   (ws + WS_CSQ);
    float*    part = (float*)   (ws + WS_PART);

    prep_b_kernel  <<<128, 256, 0, stream>>>(cent, bhi, blo);   // 32768 records
    prep_csq_kernel<<<4,   256, 0, stream>>>(cent, csq);        // 1024 clusters
    assign_kernel  <<<N_STRIPS / WAVES_PER_BLOCK, 256, 2 * BUF_BYTES, stream>>>(
        emb, bhi, blo, csq, out, part);                         // 2048 blocks
    finalize_kernel<<<1, 256, 0, stream>>>(part, out);
}
